// MultimodalGCN_27805618274576
// MI455X (gfx1250) — compile-verified, matrix operand order checked by reading, not run on hardware
//
#include <hip/hip_runtime.h>
#include <stdint.h>

// ---------------- problem constants (match reference) ----------------
#define N_NODES 100000
#define N_EDGES 1600000
#define N_BATCH 256
#define D_IN    128
#define H_DIM   256
#define BN_EPS  1e-5f

typedef __attribute__((ext_vector_type(16))) __bf16 v16bf;
typedef __attribute__((ext_vector_type(8)))  float  v8f;

union Frag { v16bf v; uint4 q[2]; };

__device__ __forceinline__ unsigned short f2bf(float f) {
  unsigned int u = __float_as_uint(f);
  u += 0x7FFFu + ((u >> 16) & 1u);           // round-to-nearest-even
  return (unsigned short)(u >> 16);
}

// ---------------- utility kernels ----------------
__global__ void fill_f32(float* __restrict__ p, long n, float v) {
  long i = (long)blockIdx.x * blockDim.x + threadIdx.x;
  if (i < n) p[i] = v;
}

__global__ void convert_to_bf16(const float* __restrict__ src,
                                unsigned short* __restrict__ dst, long n) {
  long i = (long)blockIdx.x * blockDim.x + threadIdx.x;
  if (i < n) dst[i] = f2bf(src[i]);
}

// W is [K, Nc] row-major f32 -> Bt is [Nc, K] bf16 (transposed for B-fragment loads)
__global__ void transpose_to_bf16(const float* __restrict__ W,
                                  unsigned short* __restrict__ Bt, int K, int Nc) {
  int idx = blockIdx.x * blockDim.x + threadIdx.x;
  if (idx >= K * Nc) return;
  int k = idx / Nc, n = idx % Nc;
  Bt[n * K + k] = f2bf(W[idx]);
}

// ---------------- degree / normalization ----------------
__global__ void deg_accum(const int* __restrict__ dst, const float* __restrict__ ew,
                          float* __restrict__ deg, int E) {
  int e = blockIdx.x * blockDim.x + threadIdx.x;
  if (e < E) atomicAdd(&deg[dst[e]], ew[e]);
}

__global__ void finalize_dinv(float* __restrict__ d, int n) {
  int i = blockIdx.x * blockDim.x + threadIdx.x;
  if (i >= n) return;
  float v = d[i];
  d[i] = (v > 0.0f) ? rsqrtf(v) : 0.0f;
}

// ---------------- WMMA GEMM: C[M,Nc] = A[M,K] x B[K,Nc] (bf16 in, f32 out) ----
// A: bf16 row-major [M,K]; Bt: bf16 [Nc,K] (B transposed).
// One wave computes a 16x64 strip: 4 accumulators reuse one A fragment.
// K is a template constant: full unroll + register double-buffering so the
// next step's 10 global_load_b128 are in flight while this step's WMMAs run.
// All offsets 32-bit (buffers < 4GB) -> saddr + 32-bit voffset addressing.

__device__ __forceinline__ void load_step(
    const uint4* __restrict__ A128, const uint4* __restrict__ B128,
    unsigned arow, unsigned br0, unsigned br1, unsigned br2, unsigned br3,
    unsigned kb, Frag& a, Frag& f0, Frag& f1, Frag& f2, Frag& f3)
{
  a.q[0]  = A128[(arow + kb     ) >> 3];
  a.q[1]  = A128[(arow + kb + 16) >> 3];
  f0.q[0] = B128[(br0  + kb     ) >> 3];
  f0.q[1] = B128[(br0  + kb + 16) >> 3];
  f1.q[0] = B128[(br1  + kb     ) >> 3];
  f1.q[1] = B128[(br1  + kb + 16) >> 3];
  f2.q[0] = B128[(br2  + kb     ) >> 3];
  f2.q[1] = B128[(br2  + kb + 16) >> 3];
  f3.q[0] = B128[(br3  + kb     ) >> 3];
  f3.q[1] = B128[(br3  + kb + 16) >> 3];
}

template <int K>
__global__ void __launch_bounds__(256) wmma_gemm_bf16(
    const unsigned short* __restrict__ A, const unsigned short* __restrict__ Bt,
    float* __restrict__ C, int M, int Nc)
{
  const int lane    = threadIdx.x & 31;
  const int wave    = threadIdx.x >> 5;
  const int tilesM  = (M + 15) >> 4;
  const int groupsN = Nc >> 6;                    // 64-wide column groups
  const int work    = blockIdx.x * 8 + wave;
  if (work >= tilesM * groupsN) return;           // uniform per wave
  const int tm = work / groupsN;
  const int gn = work % groupsN;

  const int half  = lane >> 4;                    // lane group 0 / 1
  const int l15   = lane & 15;
  int mrow = tm * 16 + l15; if (mrow >= M) mrow = M - 1;   // clamp: keep EXEC full
  const unsigned kbase = (unsigned)(half << 3);   // 0 or 8

  const uint4* A128 = (const uint4*)A;            // 8 bf16 per uint4
  const uint4* B128 = (const uint4*)Bt;
  const unsigned arow = (unsigned)mrow * (unsigned)K;      // bf16 element offsets
  const unsigned br0  = (unsigned)(gn * 64 +  0 + l15) * (unsigned)K;
  const unsigned br1  = (unsigned)(gn * 64 + 16 + l15) * (unsigned)K;
  const unsigned br2  = (unsigned)(gn * 64 + 32 + l15) * (unsigned)K;
  const unsigned br3  = (unsigned)(gn * 64 + 48 + l15) * (unsigned)K;

  v8f acc0 = {0.f,0.f,0.f,0.f,0.f,0.f,0.f,0.f};
  v8f acc1 = acc0, acc2 = acc0, acc3 = acc0;

  Frag a, b0, b1, b2, b3;                         // current
  load_step(A128, B128, arow, br0, br1, br2, br3, kbase, a, b0, b1, b2, b3);

#pragma unroll
  for (int kk = 0; kk < K; kk += 32) {
    Frag an, b0n, b1n, b2n, b3n;                  // next (prefetch)
    if (kk + 32 < K)
      load_step(A128, B128, arow, br0, br1, br2, br3,
                kbase + (unsigned)(kk + 32), an, b0n, b1n, b2n, b3n);
    acc0 = __builtin_amdgcn_wmma_f32_16x16x32_bf16(false, a.v, false, b0.v, (short)0, acc0, false, false);
    acc1 = __builtin_amdgcn_wmma_f32_16x16x32_bf16(false, a.v, false, b1.v, (short)0, acc1, false, false);
    acc2 = __builtin_amdgcn_wmma_f32_16x16x32_bf16(false, a.v, false, b2.v, (short)0, acc2, false, false);
    acc3 = __builtin_amdgcn_wmma_f32_16x16x32_bf16(false, a.v, false, b3.v, (short)0, acc3, false, false);
    a = an; b0 = b0n; b1 = b1n; b2 = b2n; b3 = b3n;   // renamed away by unroll
  }

  // C/D layout: VGPR r -> row (tm*16 + 8*half + r), col (gn*64 + tile*16 + l15)
  const int      mbase = tm * 16 + (half << 3);
  const unsigned ccol  = (unsigned)(gn * 64 + l15);
  if (tm * 16 + 16 <= M) {                        // full tile: unguarded stores
#pragma unroll
    for (int r = 0; r < 8; ++r) {
      float* crow = C + (unsigned)(mbase + r) * (unsigned)Nc + ccol;
      crow[ 0] = acc0[r];
      crow[16] = acc1[r];
      crow[32] = acc2[r];
      crow[48] = acc3[r];
    }
  } else {
    for (int r = 0; r < 8; ++r) {
      int m = mbase + r;
      if (m < M) {
        float* crow = C + (unsigned)m * (unsigned)Nc + ccol;
        crow[ 0] = acc0[r];
        crow[16] = acc1[r];
        crow[32] = acc2[r];
        crow[48] = acc3[r];
      }
    }
  }
}

// ---------------- edge scatter-add (one wave per edge, L2 atomics) ----------
__global__ void __launch_bounds__(256) edge_aggregate(
    const int* __restrict__ src, const int* __restrict__ dst,
    const float* __restrict__ ew, const float* __restrict__ dinv,
    const float* __restrict__ Hh, float* __restrict__ Agg, int E)
{
  int e = blockIdx.x * 8 + (threadIdx.x >> 5);
  if (e >= E) return;                            // uniform per wave
  int lane = threadIdx.x & 31;
  int s = src[e], d = dst[e];
  float norm = dinv[s] * ew[e] * dinv[d];
  const float4* hs = (const float4*)(Hh + (long)s * H_DIM);
  float* ad = Agg + (long)d * H_DIM;
#pragma unroll
  for (int j = 0; j < 2; ++j) {
    int c4 = lane + 32 * j;                      // 64 float4 = 256 features
    float4 v = hs[c4];
    int c = c4 * 4;
    atomicAdd(ad + c + 0, v.x * norm);
    atomicAdd(ad + c + 1, v.y * norm);
    atomicAdd(ad + c + 2, v.z * norm);
    atomicAdd(ad + c + 3, v.w * norm);
  }
}

// ---------------- fused self-loop + bias + BN + ReLU (in-place on Agg) ------
__global__ void fused_selfloop_bn_relu(
    float* __restrict__ Agg, const float* __restrict__ Hh,
    const float* __restrict__ dinv, const float* __restrict__ bias,
    const float* __restrict__ g, const float* __restrict__ be,
    const float* __restrict__ rm, const float* __restrict__ rv, long total)
{
  long idx = (long)blockIdx.x * blockDim.x + threadIdx.x;
  if (idx >= total) return;
  int n = (int)(idx >> 8);
  int c = (int)(idx & 255);
  float s = dinv[n];
  float v = Agg[idx] + Hh[idx] * s * s + bias[c];
  v = (v - rm[c]) * rsqrtf(rv[c] + BN_EPS) * g[c] + be[c];
  Agg[idx] = fmaxf(v, 0.0f);
}

// ---------------- pooling ----------------
__global__ void pool_count(const int* __restrict__ batch, float* __restrict__ cnt, int n) {
  int i = blockIdx.x * blockDim.x + threadIdx.x;
  if (i < n) atomicAdd(&cnt[batch[i]], 1.0f);
}

__global__ void pool_accum(const float* __restrict__ Hh, const int* __restrict__ batch,
                           float* __restrict__ embsum) {
  long idx = (long)blockIdx.x * blockDim.x + threadIdx.x;
  if (idx >= (long)N_NODES * 64) return;
  int n = (int)(idx >> 6);
  int c = (int)(idx & 63) * 4;
  int b = batch[n];
  float4 v = *(const float4*)(Hh + (long)n * H_DIM + c);
  float* dp = embsum + (long)b * H_DIM + c;
  atomicAdd(dp + 0, v.x);
  atomicAdd(dp + 1, v.y);
  atomicAdd(dp + 2, v.z);
  atomicAdd(dp + 3, v.w);
}

__global__ void emb_finalize(const float* __restrict__ embsum,
                             const float* __restrict__ cnt,
                             float* __restrict__ emb_out) {
  int idx = blockIdx.x * blockDim.x + threadIdx.x;
  if (idx >= N_BATCH * H_DIM) return;
  int b = idx >> 8;
  emb_out[idx] = embsum[idx] / fmaxf(cnt[b], 1.0f);
}

// ---------------- small classifier GEMM (+ optional BN/ReLU) ----------------
__global__ void small_gemm(const float* __restrict__ A, const float* __restrict__ W,
                           const float* __restrict__ bias,
                           const float* __restrict__ g, const float* __restrict__ be,
                           const float* __restrict__ rm, const float* __restrict__ rv,
                           float* __restrict__ out, int M, int Nc, int K, int do_bn_relu)
{
  int idx = blockIdx.x * blockDim.x + threadIdx.x;
  if (idx >= M * Nc) return;
  int m = idx / Nc, n = idx % Nc;
  float acc = bias[n];
  for (int k = 0; k < K; ++k) acc = fmaf(A[m * K + k], W[k * Nc + n], acc);
  if (do_bn_relu) {
    acc = (acc - rm[n]) * rsqrtf(rv[n] + BN_EPS) * g[n] + be[n];
    acc = fmaxf(acc, 0.0f);
  }
  out[idx] = acc;
}

// ---------------- host orchestration ----------------
extern "C" void kernel_launch(void* const* d_in, const int* in_sizes, int n_in,
                              void* d_out, int out_size, void* d_ws, size_t ws_size,
                              hipStream_t stream) {
  const float* x    = (const float*)d_in[0];
  const int*   ei   = (const int*)  d_in[1];
  const float* ew   = (const float*)d_in[2];
  const int*   bat  = (const int*)  d_in[3];
  const float* W1   = (const float*)d_in[4];
  const float* b1   = (const float*)d_in[5];
  const float* g1   = (const float*)d_in[6];
  const float* be1  = (const float*)d_in[7];
  const float* rm1  = (const float*)d_in[8];
  const float* rv1  = (const float*)d_in[9];
  const float* W2   = (const float*)d_in[10];
  const float* b2   = (const float*)d_in[11];
  const float* g2   = (const float*)d_in[12];
  const float* be2  = (const float*)d_in[13];
  const float* rm2  = (const float*)d_in[14];
  const float* rv2  = (const float*)d_in[15];
  const float* cW1  = (const float*)d_in[16];
  const float* cb1  = (const float*)d_in[17];
  const float* cg1  = (const float*)d_in[18];
  const float* cbe1 = (const float*)d_in[19];
  const float* crm1 = (const float*)d_in[20];
  const float* crv1 = (const float*)d_in[21];
  const float* cW2  = (const float*)d_in[22];
  const float* cb2  = (const float*)d_in[23];
  const float* cg2  = (const float*)d_in[24];
  const float* cbe2 = (const float*)d_in[25];
  const float* crm2 = (const float*)d_in[26];
  const float* crv2 = (const float*)d_in[27];
  const float* cW3  = (const float*)d_in[28];
  const float* cb3  = (const float*)d_in[29];

  const int* src = ei;
  const int* dst = ei + N_EDGES;

  // --- workspace carving (256B aligned) ---
  char* ws = (char*)d_ws;
  size_t off = 0;
  auto carve = [&](size_t bytes) -> char* {
    char* p = ws + off;
    off += (bytes + 255) & ~(size_t)255;
    return p;
  };
  float*          dinv   = (float*)         carve((size_t)N_NODES * 4);
  float*          Hbuf   = (float*)         carve((size_t)N_NODES * H_DIM * 4);
  float*          Abuf   = (float*)         carve((size_t)N_NODES * H_DIM * 4);
  unsigned short* Abf    = (unsigned short*)carve((size_t)N_NODES * H_DIM * 2);
  unsigned short* Wbf    = (unsigned short*)carve((size_t)H_DIM * H_DIM * 2);
  float*          embsum = (float*)         carve((size_t)N_BATCH * H_DIM * 4);
  float*          cnt    = (float*)         carve((size_t)N_BATCH * 4);
  float*          z1     = (float*)         carve((size_t)N_BATCH * 256 * 4);
  float*          z2     = (float*)         carve((size_t)N_BATCH * 128 * 4);
  (void)ws_size; (void)n_in; (void)in_sizes; (void)out_size;

  float* logits  = (float*)d_out;           // [256, 2]
  float* emb_out = (float*)d_out + 512;     // [256, 256]

  const long NH   = (long)N_NODES * H_DIM;  // 25.6M
  const int  T    = 256;
  auto blks = [](long n) { return (int)((n + 255) / 256); };
  // one wave per 16x64 strip: (M/16) * (Nc/64) work items, 8 waves per block
  const int gemmBlocks = ((N_NODES + 15) / 16 * (H_DIM / 64) + 7) / 8;   // 3125

  // --- degree / symmetric normalization ---
  fill_f32<<<blks(N_NODES), T, 0, stream>>>(dinv, N_NODES, 1.0f);  // self-loop
  deg_accum<<<blks(N_EDGES), T, 0, stream>>>(dst, ew, dinv, N_EDGES);
  finalize_dinv<<<blks(N_NODES), T, 0, stream>>>(dinv, N_NODES);

  // --- layer 1: H = x @ W1 (bf16 WMMA), aggregate, fuse BN+ReLU ---
  convert_to_bf16<<<blks((long)N_NODES * D_IN), T, 0, stream>>>(x, Abf, (long)N_NODES * D_IN);
  transpose_to_bf16<<<blks(D_IN * H_DIM), T, 0, stream>>>(W1, Wbf, D_IN, H_DIM);
  wmma_gemm_bf16<D_IN><<<gemmBlocks, T, 0, stream>>>(Abf, Wbf, Hbuf, N_NODES, H_DIM);
  fill_f32<<<blks(NH), T, 0, stream>>>(Abuf, NH, 0.0f);
  edge_aggregate<<<(N_EDGES + 7) / 8, T, 0, stream>>>(src, dst, ew, dinv, Hbuf, Abuf, N_EDGES);
  fused_selfloop_bn_relu<<<blks(NH), T, 0, stream>>>(Abuf, Hbuf, dinv, b1, g1, be1, rm1, rv1, NH);
  // Abuf now holds h_post (= residual `res`)

  // --- pool residual now (pool(final) = pool(relu2) + pool(res)) ---
  fill_f32<<<blks(N_BATCH * H_DIM), T, 0, stream>>>(embsum, N_BATCH * H_DIM, 0.0f);
  fill_f32<<<blks(N_BATCH), T, 0, stream>>>(cnt, N_BATCH, 0.0f);
  pool_count<<<blks(N_NODES), T, 0, stream>>>(bat, cnt, N_NODES);
  pool_accum<<<blks((long)N_NODES * 64), T, 0, stream>>>(Abuf, bat, embsum);

  // --- layer 2: H = h_post @ W2, aggregate, fuse BN+ReLU ---
  convert_to_bf16<<<blks(NH), T, 0, stream>>>(Abuf, Abf, NH);
  transpose_to_bf16<<<blks(H_DIM * H_DIM), T, 0, stream>>>(W2, Wbf, H_DIM, H_DIM);
  wmma_gemm_bf16<H_DIM><<<gemmBlocks, T, 0, stream>>>(Abf, Wbf, Hbuf, N_NODES, H_DIM);
  fill_f32<<<blks(NH), T, 0, stream>>>(Abuf, NH, 0.0f);   // Abuf reused as agg2
  edge_aggregate<<<(N_EDGES + 7) / 8, T, 0, stream>>>(src, dst, ew, dinv, Hbuf, Abuf, N_EDGES);
  fused_selfloop_bn_relu<<<blks(NH), T, 0, stream>>>(Abuf, Hbuf, dinv, b2, g2, be2, rm2, rv2, NH);
  pool_accum<<<blks((long)N_NODES * 64), T, 0, stream>>>(Abuf, bat, embsum);

  // --- mean pool -> emb (also an output) ---
  emb_finalize<<<blks(N_BATCH * H_DIM), T, 0, stream>>>(embsum, cnt, emb_out);

  // --- classifier MLP ---
  small_gemm<<<blks(N_BATCH * 256), T, 0, stream>>>(emb_out, cW1, cb1, cg1, cbe1, crm1, crv1,
                                                    z1, N_BATCH, 256, 256, 1);
  small_gemm<<<blks(N_BATCH * 128), T, 0, stream>>>(z1, cW2, cb2, cg2, cbe2, crm2, crv2,
                                                    z2, N_BATCH, 128, 256, 1);
  small_gemm<<<blks(N_BATCH * 2), T, 0, stream>>>(z2, cW3, cb3, nullptr, nullptr, nullptr, nullptr,
                                                  logits, N_BATCH, 2, 128, 0);
}
